// BidirectionalLSTM_73349451481725
// MI455X (gfx1250) — compile-verified, matrix operand order checked by reading, not run on hardware
//
#include <hip/hip_runtime.h>
#include <hip/hip_bf16.h>

typedef __attribute__((ext_vector_type(16))) _Float16 v16h;
typedef __attribute__((ext_vector_type(8)))  float    v8f;

#define BB 256   // batch
#define CC 128   // input channels
#define TT 256   // time steps
#define HH 256   // hidden size per direction (FH == BH)
#define KK (CC + HH)   // 384: concatenated K dim [x_t | h]
#define G4 (4 * HH)    // 1024: gate rows
#define WSTRIDE 392    // padded LDS row stride in halves (196 DW === 4 mod 64 banks)

// ---------------- workspace layout (bytes) ----------------
static constexpr size_t SZ_XT   = (size_t)TT * BB * CC * 2;  // fp16 x, time-major
static constexpr size_t SZ_W    = (size_t)G4 * KK * 2;       // fp16 packed [Wih|Whh]
static constexpr size_t SZ_BIAS = (size_t)G4 * 4;            // fp32 fused bias
static constexpr size_t SZ_H    = (size_t)BB * HH * 2;       // fp16 one h buffer
static constexpr size_t SZ_C    = (size_t)BB * HH * 4;       // fp32 c state

static constexpr size_t OFF_XT = 0;
static constexpr size_t OFF_WF = OFF_XT + SZ_XT;
static constexpr size_t OFF_WB = OFF_WF + SZ_W;
static constexpr size_t OFF_BF = OFF_WB + SZ_W;
static constexpr size_t OFF_BBi= OFF_BF + SZ_BIAS;
static constexpr size_t OFF_HF = OFF_BBi + SZ_BIAS;          // 2 buffers
static constexpr size_t OFF_HB = OFF_HF + 2 * SZ_H;          // 2 buffers
static constexpr size_t OFF_CF = OFF_HB + 2 * SZ_H;
static constexpr size_t OFF_CB = OFF_CF + SZ_C;

// ---------------- init: pack weights (fp16), fuse bias, zero states ----------------
__global__ __launch_bounds__(256) void lstm_init(
    const float* __restrict__ Wih_f, const float* __restrict__ Whh_f,
    const float* __restrict__ bih_f, const float* __restrict__ bhh_f,
    const float* __restrict__ Wih_b, const float* __restrict__ Whh_b,
    const float* __restrict__ bih_b, const float* __restrict__ bhh_b,
    _Float16* __restrict__ Wcat_f, _Float16* __restrict__ Wcat_b,
    float* __restrict__ bias_f, float* __restrict__ bias_b,
    _Float16* __restrict__ h0_f, _Float16* __restrict__ h0_b,
    float* __restrict__ c_f, float* __restrict__ c_b)
{
    const int NW  = 2 * G4 * KK;         // 786432
    const int NBI = 2 * G4;              // 2048
    const int NH  = 2 * BB * HH;         // 131072 (fp16 zeros, buffer 0 of each dir)
    const int NC  = 2 * BB * HH;         // 131072 (fp32 zeros)
    int idx = blockIdx.x * blockDim.x + threadIdx.x;

    if (idx < NW) {
        int dir = idx / (G4 * KK);
        int r   = idx % (G4 * KK);
        int g   = r / KK;
        int k   = r % KK;
        const float* Wih = dir ? Wih_b : Wih_f;
        const float* Whh = dir ? Whh_b : Whh_f;
        float v = (k < CC) ? Wih[g * CC + k] : Whh[g * HH + (k - CC)];
        (dir ? Wcat_b : Wcat_f)[(size_t)g * KK + k] = (_Float16)v;
        return;
    }
    idx -= NW;
    if (idx < NBI) {
        int dir = idx / G4, g = idx % G4;
        if (dir) bias_b[g] = bih_b[g] + bhh_b[g];
        else     bias_f[g] = bih_f[g] + bhh_f[g];
        return;
    }
    idx -= NBI;
    if (idx < NH) {
        int dir = idx / (BB * HH), e = idx % (BB * HH);
        (dir ? h0_b : h0_f)[e] = (_Float16)0.0f;
        return;
    }
    idx -= NH;
    if (idx < NC) {
        int dir = idx / (BB * HH), e = idx % (BB * HH);
        (dir ? c_b : c_f)[e] = 0.0f;
    }
}

// ---------------- x: [B,C,T] fp32 -> xT: [T,B,C] fp16 ----------------
__global__ __launch_bounds__(256) void lstm_xpose(
    const float* __restrict__ x, _Float16* __restrict__ xT)
{
    int idx = blockIdx.x * blockDim.x + threadIdx.x;  // grid sized exactly T*B*C
    int c = idx % CC;
    int b = (idx / CC) % BB;
    int t = idx / (CC * BB);
    xT[idx] = (_Float16)x[((size_t)b * CC + c) * TT + t];
}

__device__ __forceinline__ float sigm(float x) {
    return 1.0f / (1.0f + __expf(-x));
}
__device__ __forceinline__ float tanh_fast(float x) {
    return 1.0f - 2.0f / (1.0f + __expf(2.0f * x));
}

// CDNA5 async global->LDS copy of 16 bytes (CDNA5 ISA).
__device__ __forceinline__ void async_g2l_b128(unsigned lds_off, const void* gaddr) {
    asm volatile("global_load_async_to_lds_b128 %0, %1, off"
                 :: "v"(lds_off), "v"(gaddr)
                 : "memory");
}
__device__ __forceinline__ void wait_async0() {
    asm volatile("s_wait_asynccnt 0x0" ::: "memory");
}

// ---------------- one recurrent step, both directions ----------------
// Block = (dir, hidden-tile nt, batch-half); 8 waves per block cover 8 batch
// tiles and share one LDS-staged weight slab (4 gates x 16 rows x 384 halves,
// padded to WSTRIDE=392 halves/row for bank-conflict-free ds_load_b128).
// c-state and bias are prefetched while the async LDS fill is in flight.
__global__ __launch_bounds__(256) void lstm_step(
    const _Float16* __restrict__ xT,
    const _Float16* __restrict__ Wf, const _Float16* __restrict__ Wb,
    const float* __restrict__ bf, const float* __restrict__ bb,
    const _Float16* __restrict__ hin_f, _Float16* __restrict__ hout_f,
    const _Float16* __restrict__ hin_b, _Float16* __restrict__ hout_b,
    float* __restrict__ c_f, float* __restrict__ c_b,
    float* __restrict__ out, int t)
{
    __shared__ _Float16 Wlds[64 * WSTRIDE];   // 50176 bytes

    const int dir   = blockIdx.x >> 5;        // 0 = forward, 1 = backward
    const int nt    = (blockIdx.x >> 1) & 15; // hidden tile (16 cols)
    const int mhalf = blockIdx.x & 1;         // batch half
    const int mt    = mhalf * 8 + (threadIdx.x >> 5);  // batch tile
    const int lane  = threadIdx.x & 31;
    const int tt    = dir ? (TT - 1 - t) : t;

    const _Float16* W    = dir ? Wb     : Wf;
    const float*    bi   = dir ? bb     : bf;
    const _Float16* hin  = dir ? hin_b  : hin_f;
    _Float16*       hout = dir ? hout_b : hout_f;
    float*          cst  = dir ? c_b    : c_f;

    // ---- async-stage this block's 64 weight rows into padded LDS ----
    // 64 rows x 384 halves = 3072 16B chunks; 256 threads x 12 chunks.
    {
        const unsigned wbase = (unsigned)(size_t)(void*)&Wlds[0];
#pragma unroll
        for (int it = 0; it < 12; ++it) {
            int v   = it * 256 + threadIdx.x;   // 0..3071
            int row = v / 48;                   // 0..63 (gate*16 + r)
            int c16 = v % 48;                   // 16B chunk within row
            int grow = ((row >> 4) * HH) + nt * 16 + (row & 15);  // row in Wcat
            const _Float16* src = W + (size_t)grow * KK + c16 * 8;
            unsigned dst = wbase + (unsigned)(row * WSTRIDE + c16 * 8) * 2u;
            async_g2l_b128(dst, (const void*)src);
        }
    }

    const int laneRow = lane & 15;
    const int half16  = (lane >> 4) << 4;   // 0 for lanes 0-15, 16 for lanes 16-31

    // A-matrix row (batch) this lane supplies
    const int rowA = mt * 16 + laneRow;
    const _Float16* xrow = xT + ((size_t)tt * BB + rowA) * CC;
    const _Float16* hrow = hin + (size_t)rowA * HH;

    // ---- prefetch epilogue operands while the async fill is in flight ----
    const int hid = nt * 16 + laneRow;
    const float bv0 = bi[0 * HH + hid];
    const float bv1 = bi[1 * HH + hid];
    const float bv2 = bi[2 * HH + hid];
    const float bv3 = bi[3 * HH + hid];
    const int rowBase = mt * 16 + ((lane >> 4) << 3);  // +8 for upper half-lanes
    float cold[8];
#pragma unroll
    for (int j = 0; j < 8; ++j)
        cold[j] = cst[(size_t)(rowBase + j) * HH + hid];

    wait_async0();
    __syncthreads();

    v8f acc0 = {}, acc1 = {}, acc2 = {}, acc3 = {};

#pragma unroll
    for (int kt = 0; kt < KK / 32; ++kt) {          // 12 k-tiles of 32
        const int kh = kt * 32 + half16;
        // kt<4 -> x part (kh+15 <= 127); kt>=4 -> h part
        const _Float16* aptr = (kt < 4) ? (xrow + kh) : (hrow + (kh - CC));
        v16h a  = *(const v16h*)aptr;
        v16h b0 = *(const v16h*)&Wlds[( 0 + laneRow) * WSTRIDE + kh];
        v16h b1 = *(const v16h*)&Wlds[(16 + laneRow) * WSTRIDE + kh];
        v16h b2 = *(const v16h*)&Wlds[(32 + laneRow) * WSTRIDE + kh];
        v16h b3 = *(const v16h*)&Wlds[(48 + laneRow) * WSTRIDE + kh];
        acc0 = __builtin_amdgcn_wmma_f32_16x16x32_f16(false, a, false, b0, (short)0, acc0, false, false);
        acc1 = __builtin_amdgcn_wmma_f32_16x16x32_f16(false, a, false, b1, (short)0, acc1, false, false);
        acc2 = __builtin_amdgcn_wmma_f32_16x16x32_f16(false, a, false, b2, (short)0, acc2, false, false);
        acc3 = __builtin_amdgcn_wmma_f32_16x16x32_f16(false, a, false, b3, (short)0, acc3, false, false);
    }

    // ---- epilogue: bias, activations, c/h update, output store ----
#pragma unroll
    for (int j = 0; j < 8; ++j) {
        float gi = acc0[j] + bv0;
        float gf = acc1[j] + bv1;
        float gg = acc2[j] + bv2;
        float go = acc3[j] + bv3;
        float i_ = sigm(gi);
        float f_ = sigm(gf);
        float g_ = tanh_fast(gg);
        float o_ = sigm(go);

        const int row = rowBase + j;                 // batch index
        const size_t sidx = (size_t)row * HH + hid;  // state index
        float cn = f_ * cold[j] + i_ * g_;
        cst[sidx] = cn;
        float hv = o_ * tanh_fast(cn);
        hout[sidx] = (_Float16)hv;
        // out: [B, 2H, T], forward -> channels [0,H), backward -> [H,2H)
        out[((size_t)row * (2 * HH) + (size_t)dir * HH + hid) * TT + tt] = hv;
    }
}

// ---------------- host side ----------------
extern "C" void kernel_launch(void* const* d_in, const int* in_sizes, int n_in,
                              void* d_out, int out_size, void* d_ws, size_t ws_size,
                              hipStream_t stream) {
    const float* x     = (const float*)d_in[0];
    const float* Wih_f = (const float*)d_in[1];
    const float* Whh_f = (const float*)d_in[2];
    const float* bih_f = (const float*)d_in[3];
    const float* bhh_f = (const float*)d_in[4];
    const float* Wih_b = (const float*)d_in[5];
    const float* Whh_b = (const float*)d_in[6];
    const float* bih_b = (const float*)d_in[7];
    const float* bhh_b = (const float*)d_in[8];

    char* ws = (char*)d_ws;
    _Float16* xT     = (_Float16*)(ws + OFF_XT);
    _Float16* Wcat_f = (_Float16*)(ws + OFF_WF);
    _Float16* Wcat_b = (_Float16*)(ws + OFF_WB);
    float*    bias_f = (float*)(ws + OFF_BF);
    float*    bias_b = (float*)(ws + OFF_BBi);
    _Float16* hF     = (_Float16*)(ws + OFF_HF);  // 2 buffers of BB*HH
    _Float16* hB     = (_Float16*)(ws + OFF_HB);  // 2 buffers of BB*HH
    float*    c_f    = (float*)(ws + OFF_CF);
    float*    c_b    = (float*)(ws + OFF_CB);
    float*    out    = (float*)d_out;

    // init: weights + bias + zero h0/c
    {
        int total = 2 * G4 * KK + 2 * G4 + 2 * BB * HH + 2 * BB * HH;
        lstm_init<<<(total + 255) / 256, 256, 0, stream>>>(
            Wih_f, Whh_f, bih_f, bhh_f, Wih_b, Whh_b, bih_b, bhh_b,
            Wcat_f, Wcat_b, bias_f, bias_b,
            hF /*buffer 0*/, hB /*buffer 0*/, c_f, c_b);
    }
    // transpose/convert x
    {
        int total = TT * BB * CC;  // 8388608, multiple of 256
        lstm_xpose<<<total / 256, 256, 0, stream>>>(x, xT);
    }
    // recurrent scan: 256 stream-ordered step launches, h ping-pong by parity
    for (int t = 0; t < TT; ++t) {
        const _Float16* hinf  = hF + (size_t)(t & 1) * BB * HH;
        _Float16*       houtf = hF + (size_t)((t + 1) & 1) * BB * HH;
        const _Float16* hinb  = hB + (size_t)(t & 1) * BB * HH;
        _Float16*       houtb = hB + (size_t)((t + 1) & 1) * BB * HH;
        // 64 blocks x 256 threads (8 waves): block = (dir, hidden tile, batch half)
        lstm_step<<<64, 256, 0, stream>>>(
            xT, Wcat_f, Wcat_b, bias_f, bias_b,
            hinf, houtf, hinb, houtb, c_f, c_b, out, t);
    }
}
